// SNNIDEBERTSpikingMultiLayerModule_83262236000907
// MI455X (gfx1250) — matrix operand order repaired and checked by reading, not compiled, not run on hardware
//
#include <hip/hip_runtime.h>
#include <hip/hip_bf16.h>

// ---------------- problem constants (match reference) ----------------
#define NB    4      // batch
#define SEQ   256    // sequence length
#define BDIM  768    // model dim D
#define NH    12     // heads
#define DHD   64     // head dim
#define FDIM  3072   // FFN dim
#define NL    4      // layers
#define TSTEPS 6     // time_step (reference constant)

typedef __attribute__((ext_vector_type(16))) _Float16 v16h;
typedef __attribute__((ext_vector_type(8)))  float    v8f;

__device__ __forceinline__ float gelu_tanh(float x) {
    float x3 = x * x * x;
    return 0.5f * x * (1.0f + tanhf(0.7978845608028654f * (x + 0.044715f * x3)));
}

union h4pack { _Float16 h[4]; uint2 u2; };

__device__ __forceinline__ void pack_store4(_Float16* dst, float x, float y, float z, float w) {
    h4pack p;
    p.h[0] = (_Float16)x; p.h[1] = (_Float16)y;
    p.h[2] = (_Float16)z; p.h[3] = (_Float16)w;
    *(uint2*)dst = p.u2;   // 8B-aligned -> ds_store_b64
}

// ---------------------------------------------------------------------
// Batched GEMM: C = alpha * A(f32->f16) x B(f32->f16) + bias, optional GELU.
// Block tile 128x64, K-chunk 32, LDS double-buffered (one barrier / chunk).
// 256 threads = 8 waves, wave grid 4(M)x2(N), each wave computes a 32x32
// tile via 2x2 v_wmma_f32_16x16x32_f16.
//
// LDS layouts chosen so every WMMA fragment is contiguous, 16B-aligned
// runs -> ds_load_b128:
//   As[buf][m][k] (row pitch 40 halfs = 80B): A frag = runs at k = hlf*8(+16)
//   Bs[buf][n][k] (row pitch 40 halfs = 80B): B frag = runs at k = hlf*8(+16)
// Pipeline: prefetch chunk k+32 into registers while WMMAs consume chunk k
// from buffer cur; store prefetch into buffer cur^1; single __syncthreads.
//
// Batch index z decomposed as (z/bH, z%bH) with independent strides, so the
// same kernel covers plain GEMM, per-(b,h) attention scores and prob*V.
// Dims assumed multiples of tile sizes (true for all uses here).
// ---------------------------------------------------------------------
template<int EPI /*0=none,1=gelu*/, int TRANSB>
__global__ __launch_bounds__(256) void gemm_wmma_kernel(
    const float* __restrict__ A, int lda,
    const float* __restrict__ Bw, int ldb,
    const float* __restrict__ bias,
    float* __restrict__ C, int ldc,
    int M, int N, int K, float alpha,
    int bH,
    long long sA1, long long sA2,
    long long sB1, long long sB2,
    long long sC1, long long sC2)
{
    __shared__ __align__(16) _Float16 As[2][128][40];  // 2 x 128(m) x 32(k) (+pad)
    __shared__ __align__(16) _Float16 Bs[2][64][40];   // 2 x 64(n)  x 32(k) (+pad)
    const int ABUF = 128 * 40;
    const int BBUF = 64 * 40;

    const int bz  = blockIdx.z;
    const int bzo = bz / bH;
    const int bzi = bz % bH;
    A  += bzo * sA1 + bzi * sA2;
    Bw += bzo * sB1 + bzi * sB2;
    C  += bzo * sC1 + bzi * sC2;

    const int m0   = blockIdx.y * 128;
    const int n0   = blockIdx.x * 64;
    const int tid  = threadIdx.x;
    const int wave = tid >> 5;
    const int lane = tid & 31;
    const int wm   = (wave & 3) * 32;   // wave row offset in block tile
    const int wn   = (wave >> 2) * 32;  // wave col offset in block tile
    const int hlf  = lane >> 4;         // lane group (0: lanes0-15, 1: lanes16-31)
    const int l16  = lane & 15;

    // ---- staging addresses (computed once) ----
    // A: thread -> row ar (+32*j), cols ac..ac+3
    const int ar = tid >> 3;
    const int ac = (tid & 7) << 2;
    const float*    Ap  = A + (long long)(m0 + ar) * lda + ac;
    _Float16*       AsP = &As[0][ar][ac];
    const long long ldA32 = (long long)32 * lda;

    // B (TRANSB=1, Bw[N][K]): thread -> col bc (+32*j), k bk4t..bk4t+3
    const int bc   = tid >> 3;
    const int bk4t = (tid & 7) << 2;
    const float*    BpT  = Bw + (long long)(n0 + bc) * ldb + bk4t;
    _Float16*       BsPT = &Bs[0][bc][bk4t];
    const long long ldB32 = (long long)32 * ldb;

    // B (TRANSB=0, Bw[K][N]): thread -> col bn, k group bk4 (+16*j)
    const int bn  = tid & 63;
    const int bk4 = (tid >> 6) << 2;     // 0,4,8,12
    const float* Bbase = Bw + n0 + bn;   // + k*ldb applied per row
    _Float16*    Bs0P  = &Bs[0][bn][bk4];

    v8f acc00 = {}, acc01 = {}, acc10 = {}, acc11 = {};

    float4 fa[4];
    float4 fbT[2];
    float  fb0v[2][4];

    // ---- prologue: stage chunk 0 into buffer 0 ----
#pragma unroll
    for (int j = 0; j < 4; ++j)
        fa[j] = *(const float4*)(Ap + j * ldA32);
    if (TRANSB) {
#pragma unroll
        for (int j = 0; j < 2; ++j)
            fbT[j] = *(const float4*)(BpT + j * ldB32);
    } else {
#pragma unroll
        for (int j = 0; j < 2; ++j) {
            long long krow = (long long)(j * 16 + bk4) * ldb;
#pragma unroll
            for (int kk = 0; kk < 4; ++kk)
                fb0v[j][kk] = Bbase[krow + (long long)kk * ldb];
        }
    }
#pragma unroll
    for (int j = 0; j < 4; ++j)
        pack_store4(AsP + j * 32 * 40, fa[j].x, fa[j].y, fa[j].z, fa[j].w);
    if (TRANSB) {
#pragma unroll
        for (int j = 0; j < 2; ++j)
            pack_store4(BsPT + j * 32 * 40, fbT[j].x, fbT[j].y, fbT[j].z, fbT[j].w);
    } else {
#pragma unroll
        for (int j = 0; j < 2; ++j)
            pack_store4(Bs0P + j * 16, fb0v[j][0], fb0v[j][1], fb0v[j][2], fb0v[j][3]);
    }
    __syncthreads();

    int cur = 0;
    for (int k0 = 0; k0 < K; k0 += 32) {
        const bool hasNext = (k0 + 32) < K;
        // ---- prefetch next chunk into registers (overlaps with WMMAs) ----
        if (hasNext) {
            const int kn = k0 + 32;
#pragma unroll
            for (int j = 0; j < 4; ++j)
                fa[j] = *(const float4*)(Ap + kn + j * ldA32);
            if (TRANSB) {
#pragma unroll
                for (int j = 0; j < 2; ++j)
                    fbT[j] = *(const float4*)(BpT + kn + j * ldB32);
            } else {
#pragma unroll
                for (int j = 0; j < 2; ++j) {
                    long long krow = (long long)(kn + j * 16 + bk4) * ldb;
#pragma unroll
                    for (int kk = 0; kk < 4; ++kk)
                        fb0v[j][kk] = Bbase[krow + (long long)kk * ldb];
                }
            }
        }

        // ---- gather fragments from buffer `cur` (ds_load_b128 runs) ----
        v16h a0 = {}, a1 = {}, b0 = {}, b1 = {};
        {
            const _Float16* Ab  = &As[cur][0][0];
            const _Float16* Bb  = &Bs[cur][0][0];
            const _Float16* Ar0 = Ab + (wm + l16) * 40;
            const _Float16* Ar1 = Ab + (wm + 16 + l16) * 40;
            const _Float16* Bc0 = Bb + (wn + l16) * 40;
            const _Float16* Bc1 = Bb + (wn + 16 + l16) * 40;
#pragma unroll
            for (int e = 0; e < 8; ++e) {
                a0[e]     = Ar0[hlf * 8 + e];
                a0[8 + e] = Ar0[16 + hlf * 8 + e];
                a1[e]     = Ar1[hlf * 8 + e];
                a1[8 + e] = Ar1[16 + hlf * 8 + e];
            }
#pragma unroll
            for (int e = 0; e < 8; ++e) {
                b0[e]     = Bc0[hlf * 8 + e];
                b0[8 + e] = Bc0[16 + hlf * 8 + e];
                b1[e]     = Bc1[hlf * 8 + e];
                b1[8 + e] = Bc1[16 + hlf * 8 + e];
            }
        }

        acc00 = __builtin_amdgcn_wmma_f32_16x16x32_f16(false, a0, false, b0, (short)0, acc00, false, false);
        acc01 = __builtin_amdgcn_wmma_f32_16x16x32_f16(false, a0, false, b1, (short)0, acc01, false, false);
        acc10 = __builtin_amdgcn_wmma_f32_16x16x32_f16(false, a1, false, b0, (short)0, acc10, false, false);
        acc11 = __builtin_amdgcn_wmma_f32_16x16x32_f16(false, a1, false, b1, (short)0, acc11, false, false);

        // ---- store prefetched chunk into the other buffer ----
        if (hasNext) {
            const int nb   = cur ^ 1;
            _Float16* AsPn = AsP + nb * ABUF;
#pragma unroll
            for (int j = 0; j < 4; ++j)
                pack_store4(AsPn + j * 32 * 40, fa[j].x, fa[j].y, fa[j].z, fa[j].w);
            if (TRANSB) {
                _Float16* BsPn = BsPT + nb * BBUF;
#pragma unroll
                for (int j = 0; j < 2; ++j)
                    pack_store4(BsPn + j * 32 * 40, fbT[j].x, fbT[j].y, fbT[j].z, fbT[j].w);
            } else {
                _Float16* BsPn = Bs0P + nb * BBUF;
#pragma unroll
                for (int j = 0; j < 2; ++j)
                    pack_store4(BsPn + j * 16, fb0v[j][0], fb0v[j][1], fb0v[j][2], fb0v[j][3]);
            }
        }
        __syncthreads();
        cur ^= 1;
    }

    // C/D layout: lane col N = lane%16; VGPR r -> row M = r + 8*(lane/16)
    const int col0 = n0 + wn + l16;
    const int col1 = col0 + 16;
    const float bv0 = bias ? bias[col0] : 0.0f;
    const float bv1 = bias ? bias[col1] : 0.0f;
#pragma unroll
    for (int r = 0; r < 8; ++r) {
        int row0 = m0 + wm + r + hlf * 8;
        int row1 = row0 + 16;
        float v00 = acc00[r] * alpha + bv0, v01 = acc01[r] * alpha + bv1;
        float v10 = acc10[r] * alpha + bv0, v11 = acc11[r] * alpha + bv1;
        if (EPI == 1) {
            v00 = gelu_tanh(v00); v01 = gelu_tanh(v01);
            v10 = gelu_tanh(v10); v11 = gelu_tanh(v11);
        }
        C[(long long)row0 * ldc + col0] = v00;
        C[(long long)row0 * ldc + col1] = v01;
        C[(long long)row1 * ldc + col0] = v10;
        C[(long long)row1 * ldc + col1] = v11;
    }
}

// ------------- elementwise / reduction helper kernels -------------

__global__ void embed_kernel(const float* __restrict__ u, const int* __restrict__ seg,
                             const float* __restrict__ semb,
                             float* __restrict__ x_in, float* __restrict__ z0, int n) {
    int i = blockIdx.x * blockDim.x + threadIdx.x;
    if (i >= n) return;
    int d  = i % BDIM;
    int bs = i / BDIM;
    float v = u[i] + semb[seg[bs] * BDIM + d];
    x_in[i] = v;
    z0[i]   = v;
}

__global__ void fill0_kernel(float* __restrict__ p, long long n) {
    long long i  = (long long)blockIdx.x * blockDim.x + threadIdx.x;
    long long st = (long long)gridDim.x * blockDim.x;
    for (; i < n; i += st) p[i] = 0.0f;
}

// softmax over last dim (S=256, one block per row) with BERT extended mask,
// fused accumulation of probs into the attn output slab.
__global__ __launch_bounds__(256) void softmax_mask_accum_kernel(
    float* __restrict__ scores, const float* __restrict__ mask,
    float* __restrict__ attn_acc) {
    __shared__ float red[256];
    const int row = blockIdx.x;            // (b*NH + h)*SEQ + s
    const int b   = row / (NH * SEQ);
    const int t   = threadIdx.x;
    long long idx = (long long)row * SEQ + t;
    float val = scores[idx] + (1.0f - mask[b * SEQ + t]) * (-10000.0f);
    red[t] = val; __syncthreads();
    for (int s2 = 128; s2 > 0; s2 >>= 1) { if (t < s2) red[t] = fmaxf(red[t], red[t + s2]); __syncthreads(); }
    float mx = red[0]; __syncthreads();
    float e = __expf(val - mx);
    red[t] = e; __syncthreads();
    for (int s2 = 128; s2 > 0; s2 >>= 1) { if (t < s2) red[t] += red[t + s2]; __syncthreads(); }
    float p = e / red[0];
    scores[idx]    = p;
    attn_acc[idx] += p;
}

// out = LN(x + dx) with gamma/beta; one block per row of D=768 (3 elems/thread)
__global__ __launch_bounds__(256) void add_ln_kernel(
    const float* __restrict__ x, const float* __restrict__ dx,
    const float* __restrict__ g, const float* __restrict__ b,
    float* __restrict__ out) {
    __shared__ float red[256];
    const long long row = blockIdx.x;
    const int t = threadIdx.x;
    long long base = row * BDIM;
    float v0 = x[base + t]       + dx[base + t];
    float v1 = x[base + t + 256] + dx[base + t + 256];
    float v2 = x[base + t + 512] + dx[base + t + 512];
    red[t] = v0 + v1 + v2; __syncthreads();
    for (int s2 = 128; s2 > 0; s2 >>= 1) { if (t < s2) red[t] += red[t + s2]; __syncthreads(); }
    float mean = red[0] * (1.0f / BDIM); __syncthreads();
    float d0 = v0 - mean, d1 = v1 - mean, d2 = v2 - mean;
    red[t] = d0 * d0 + d1 * d1 + d2 * d2; __syncthreads();
    for (int s2 = 128; s2 > 0; s2 >>= 1) { if (t < s2) red[t] += red[t + s2]; __syncthreads(); }
    float rs = rsqrtf(red[0] * (1.0f / BDIM) + 1e-12f);
    out[base + t]       = g[t]       * d0 * rs + b[t];
    out[base + t + 256] = g[t + 256] * d1 * rs + b[t + 256];
    out[base + t + 512] = g[t + 512] * d2 * rs + b[t + 512];
}

// IF neuron: integrate, spike, soft-reset, update count, emit running rate
__global__ void snn_update_kernel(const float* __restrict__ cur,
                                  float* __restrict__ mem, float* __restrict__ cnt,
                                  float* __restrict__ prev, int n, float inv_t1) {
    int i = blockIdx.x * blockDim.x + threadIdx.x;
    if (i >= n) return;
    float m = mem[i] + cur[i];
    float s = (m >= 1.0f) ? 1.0f : 0.0f;   // VTH = 1
    mem[i] = m - s;
    float c = cnt[i] + s;
    cnt[i]  = c;
    prev[i] = c * inv_t1;
}

__global__ void scale_copy_kernel(const float* __restrict__ in, float* __restrict__ out,
                                  long long n, float s) {
    long long i  = (long long)blockIdx.x * blockDim.x + threadIdx.x;
    long long st = (long long)gridDim.x * blockDim.x;
    for (; i < n; i += st) out[i] = in[i] * s;
}

__global__ void scale_inplace_kernel(float* __restrict__ p, long long n, float s) {
    long long i  = (long long)blockIdx.x * blockDim.x + threadIdx.x;
    long long st = (long long)gridDim.x * blockDim.x;
    for (; i < n; i += st) p[i] *= s;
}

// --------------------------- launcher ---------------------------

extern "C" void kernel_launch(void* const* d_in, const int* in_sizes, int n_in,
                              void* d_out, int out_size, void* d_ws, size_t ws_size,
                              hipStream_t stream) {
    const float* u    = (const float*)d_in[0];
    const int*   seg  = (const int*)  d_in[1];
    const float* mask = (const float*)d_in[2];
    const float* semb = (const float*)d_in[3];
    const float* Wq   = (const float*)d_in[4];
    const float* bq   = (const float*)d_in[5];
    const float* Wk   = (const float*)d_in[6];
    const float* bk   = (const float*)d_in[7];
    const float* Wv   = (const float*)d_in[8];
    const float* bv   = (const float*)d_in[9];
    const float* Wo   = (const float*)d_in[10];
    const float* bo   = (const float*)d_in[11];
    const float* ln1g = (const float*)d_in[12];
    const float* ln1b = (const float*)d_in[13];
    const float* W1   = (const float*)d_in[14];
    const float* b1   = (const float*)d_in[15];
    const float* W2   = (const float*)d_in[16];
    const float* b2   = (const float*)d_in[17];
    const float* ln2g = (const float*)d_in[18];
    const float* ln2b = (const float*)d_in[19];
    (void)in_sizes; (void)n_in; (void)out_size; (void)ws_size;

    const long long N1   = (long long)NB * SEQ * BDIM;    // 786432  (one [B,S,D])
    const long long BHSS = (long long)NB * NH * SEQ * SEQ;// 3145728 (one [B,H,S,S])
    const long long NHID = (long long)NB * SEQ * FDIM;    // 3145728 (one [B,S,F])
    const long long D2   = (long long)BDIM * BDIM;
    const long long DF   = (long long)BDIM * FDIM;

    float* ws    = (float*)d_ws;
    float* x_in  = ws; ws += N1;
    float* prev  = ws; ws += N1;
    float* qb    = ws; ws += N1;     // Q, later reused for out-proj result
    float* kb    = ws; ws += N1;     // K, later reused for x1
    float* vb    = ws; ws += N1;     // V, later reused for FFN2 result
    float* ctx   = ws; ws += N1;     // ctx, later reused for cur
    float* hbuf  = ws; ws += NHID;   // FFN hidden
    float* sc    = ws; ws += BHSS;   // scores / probs
    float* memb  = ws; ws += (long long)NL * N1;  // membrane
    float* cnt   = ws; ws += (long long)NL * N1;  // spike counts

    float* z_out    = (float*)d_out;            // [5,B,S,D]
    float* attn_out = z_out + 5 * N1;           // [L,B,H,S,S]

    dim3 b256(256);

    // x_in = u + seg_emb[segment_ids]; z[0] = x_in
    embed_kernel<<<(unsigned)((N1 + 255) / 256), b256, 0, stream>>>(u, seg, semb, x_in, z_out, (int)N1);
    // zero persistent state + attn accumulator (d_out is poisoned by harness)
    fill0_kernel<<<1024, b256, 0, stream>>>(memb, 2LL * NL * N1);
    fill0_kernel<<<1024, b256, 0, stream>>>(attn_out, (long long)NL * BHSS);

    const dim3 g_d (BDIM / 64, (NB * SEQ) / 128, 1);        // [1024 x 768] GEMMs
    const dim3 g_sc(SEQ / 64,  SEQ / 128, NB * NH);         // scores
    const dim3 g_cx(DHD / 64,  SEQ / 128, NB * NH);         // probs x V
    const dim3 g_f1(FDIM / 64, (NB * SEQ) / 128, 1);        // FFN1

    for (int t = 0; t < TSTEPS; ++t) {
        hipMemcpyAsync(prev, x_in, N1 * sizeof(float), hipMemcpyDeviceToDevice, stream);
        const float inv_t1 = 1.0f / (float)(t + 1);
        for (int l = 0; l < NL; ++l) {
            // --- QKV projections ---
            gemm_wmma_kernel<0, 0><<<g_d, b256, 0, stream>>>(
                prev, BDIM, Wq + l * D2, BDIM, bq + l * BDIM, qb, BDIM,
                NB * SEQ, BDIM, BDIM, 1.0f, 1, 0, 0, 0, 0, 0, 0);
            gemm_wmma_kernel<0, 0><<<g_d, b256, 0, stream>>>(
                prev, BDIM, Wk + l * D2, BDIM, bk + l * BDIM, kb, BDIM,
                NB * SEQ, BDIM, BDIM, 1.0f, 1, 0, 0, 0, 0, 0, 0);
            gemm_wmma_kernel<0, 0><<<g_d, b256, 0, stream>>>(
                prev, BDIM, Wv + l * D2, BDIM, bv + l * BDIM, vb, BDIM,
                NB * SEQ, BDIM, BDIM, 1.0f, 1, 0, 0, 0, 0, 0, 0);
            // --- scores = Q K^T / sqrt(DH) (batched over b,h) ---
            gemm_wmma_kernel<0, 1><<<g_sc, b256, 0, stream>>>(
                qb, BDIM, kb, BDIM, nullptr, sc, SEQ,
                SEQ, SEQ, DHD, 0.125f, NH,
                (long long)SEQ * BDIM, DHD,
                (long long)SEQ * BDIM, DHD,
                (long long)NH * SEQ * SEQ, (long long)SEQ * SEQ);
            // --- softmax (+mask) and accumulate probs into attn output ---
            softmax_mask_accum_kernel<<<NB * NH * SEQ, b256, 0, stream>>>(
                sc, mask, attn_out + (long long)l * BHSS);
            // --- ctx = P V (batched over b,h) ---
            gemm_wmma_kernel<0, 0><<<g_cx, b256, 0, stream>>>(
                sc, SEQ, vb, BDIM, nullptr, ctx, BDIM,
                SEQ, DHD, SEQ, 1.0f, NH,
                (long long)NH * SEQ * SEQ, (long long)SEQ * SEQ,
                (long long)SEQ * BDIM, DHD,
                (long long)SEQ * BDIM, DHD);
            // --- out projection -> qb ---
            gemm_wmma_kernel<0, 0><<<g_d, b256, 0, stream>>>(
                ctx, BDIM, Wo + l * D2, BDIM, bo + l * BDIM, qb, BDIM,
                NB * SEQ, BDIM, BDIM, 1.0f, 1, 0, 0, 0, 0, 0, 0);
            // --- x1 = LN(prev + attn_out) -> kb ---
            add_ln_kernel<<<NB * SEQ, b256, 0, stream>>>(
                prev, qb, ln1g + l * BDIM, ln1b + l * BDIM, kb);
            // --- h = gelu(x1 W1 + b1) -> hbuf ---
            gemm_wmma_kernel<1, 0><<<g_f1, b256, 0, stream>>>(
                kb, BDIM, W1 + l * DF, FDIM, b1 + l * FDIM, hbuf, FDIM,
                NB * SEQ, FDIM, BDIM, 1.0f, 1, 0, 0, 0, 0, 0, 0);
            // --- f = h W2 + b2 -> vb ---
            gemm_wmma_kernel<0, 0><<<g_d, b256, 0, stream>>>(
                hbuf, FDIM, W2 + l * DF, BDIM, b2 + l * BDIM, vb, BDIM,
                NB * SEQ, BDIM, FDIM, 1.0f, 1, 0, 0, 0, 0, 0, 0);
            // --- cur = LN(x1 + f) -> ctx ---
            add_ln_kernel<<<NB * SEQ, b256, 0, stream>>>(
                kb, vb, ln2g + l * BDIM, ln2b + l * BDIM, ctx);
            // --- IF neuron update; prev becomes next block's input rate ---
            snn_update_kernel<<<(unsigned)((N1 + 255) / 256), b256, 0, stream>>>(
                ctx, memb + (long long)l * N1, cnt + (long long)l * N1, prev, (int)N1, inv_t1);
        }
    }

    // z[1..4] = cnt * VTH / T ; attn = asum / T
    scale_copy_kernel<<<1024, b256, 0, stream>>>(cnt, z_out + N1, (long long)NL * N1, 1.0f / (float)TSTEPS);
    scale_inplace_kernel<<<1024, b256, 0, stream>>>(attn_out, (long long)NL * BHSS, 1.0f / (float)TSTEPS);
}
